// SoftDecisionTreeEnsemble_50062138802303
// MI455X (gfx1250) — matrix operand order repaired and checked by reading, not compiled, not run on hardware
//
#include <hip/hip_runtime.h>
#include <hip/hip_bf16.h>

typedef __attribute__((ext_vector_type(16))) __bf16 v16bf;
typedef __attribute__((ext_vector_type(8)))  float  v8f;

#define N_TREES    15
#define INPUT_DIM  128
#define N_CLASSES  10
#define N_LEAVES   8
#define N_INTERNAL 7
#define PER_TREE   (N_INTERNAL*INPUT_DIM + N_INTERNAL + N_LEAVES*N_CLASSES) /* 983 */
#define W_END      (N_INTERNAL*INPUT_DIM)                                   /* 896 */
#define LL_OFF     (W_END + N_INTERNAL)                                     /* 903 */

// GEMM1: 105 real columns (15 trees x 7 nodes), padded to 112 = 7 chunks of 16.
#define NCOLS      105
#define NCHUNK     7
#define KSTEP      4      // 128 / 32
// GEMM2: K = 15 trees * 8 leaves = 120, padded to 128; N = 10 classes pad 16.
#define K2         120
#define K2STEP     4

// ---- d_ws layout (bytes) ----
// Bpack:  [chunk(7)][kb(4)][lane(32)][16 bf16] = 28672 B
// bias:   112 f32 (pad cols -> 0)              = 448 B
// Mpack2: [kb(4)][lane(32)][16 bf16]           = 4096 B
#define WS_BPACK   0
#define WS_BIAS    28672
#define WS_MPACK2  29120
#define WS_TOTAL   33216

// ---- LDS layout (bytes) ----
#define LDS_BPACK  0
#define LDS_BIAS   28672
#define LDS_MPACK2 29120
#define LDS_Z      33216                 // 64B aligned (33216 = 64*519)
#define ZROW       124                   // f32 per row, anti-bank-conflict stride
#define ZSLAB      (32*ZROW*4)           // 15872 B per wave
#define LPROW      272                   // bytes per leaf-prob row (136 bf16; 68 dwords)
#define NWAVES     4
#define LDS_TOTAL  (LDS_Z + NWAVES*ZSLAB)  // 96704 B

__device__ __forceinline__ float fast_sigmoid(float z) {
  return __builtin_amdgcn_rcpf(1.0f + __expf(-z));
}

__device__ __forceinline__ unsigned pack_bf16(float a, float b) {
  union { __bf16 h[2]; unsigned u; } cv;
  cv.h[0] = (__bf16)a; cv.h[1] = (__bf16)b;
  return cv.u;   // -> v_cvt_pk_bf16_f32
}

// ---------------------------------------------------------------------------
// Prep: fold tree-weight softmax into leaf-class softmax, pack it into the
// wave32 bf16 B layout (Mpack2); extract bias; pack split_w (Bpack).
// ---------------------------------------------------------------------------
__global__ void sdte_prep_kernel(const float* __restrict__ params,
                                 char* __restrict__ ws) {
  __shared__ float Msh[N_TREES * N_LEAVES * N_CLASSES];   // [t][l][c], 4800 B
  __bf16* bpack  = (__bf16*)(ws + WS_BPACK);
  float*  bias   = (float*)(ws + WS_BIAS);
  __bf16* mpack  = (__bf16*)(ws + WS_MPACK2);
  const int tid = threadIdx.x;

  // Msh[t][l][c] = softmax(leaf_logits[t][l])[c] * softmax(tree_logits)[t]
  if (tid < N_TREES * N_LEAVES) {
    const int t = tid >> 3, l = tid & 7;
    const float* twl = params + N_TREES * PER_TREE;
    float mx = twl[0];
    #pragma unroll
    for (int i = 1; i < N_TREES; ++i) mx = fmaxf(mx, twl[i]);
    float s = 0.f;
    #pragma unroll
    for (int i = 0; i < N_TREES; ++i) s += __expf(twl[i] - mx);
    const float tw = __expf(twl[t] - mx) / s;

    const float* ll = params + t * PER_TREE + LL_OFF + l * N_CLASSES;
    float m2 = ll[0];
    #pragma unroll
    for (int c = 1; c < N_CLASSES; ++c) m2 = fmaxf(m2, ll[c]);
    float e[N_CLASSES]; float se = 0.f;
    #pragma unroll
    for (int c = 0; c < N_CLASSES; ++c) { e[c] = __expf(ll[c] - m2); se += e[c]; }
    const float inv = tw / se;
    #pragma unroll
    for (int c = 0; c < N_CLASSES; ++c)
      Msh[(t * N_LEAVES + l) * N_CLASSES + c] = e[c] * inv;
  }

  // bias (flat n = 7t + node), pad to 112 with zeros
  if (tid < 112) {
    float b = 0.f;
    if (tid < NCOLS) b = params[(tid / 7) * PER_TREE + W_END + (tid % 7)];
    bias[tid] = b;
  }

  // Bpack[((j*4+kb)*32 + L)*16 + 2v + s] = W[n][k], bf16
  for (int idx = tid; idx < NCHUNK * KSTEP * 32 * 16; idx += blockDim.x) {
    const int s  =  idx        & 1;
    const int v  = (idx >> 1)  & 7;
    const int L  = (idx >> 4)  & 31;
    const int kb = (idx >> 9)  & 3;
    const int j  =  idx >> 11;
    const int n  = 16 * j + (L & 15);
    const int k  = 32 * kb + ((L >= 16) ? 8 : 0) + ((v >= 4) ? 8 : 0) + 2 * v + s;
    float val = 0.f;
    if (n < NCOLS) {
      const int t = n / 7, node = n % 7;
      val = params[t * PER_TREE + node * INPUT_DIM + k];
    }
    bpack[idx] = (__bf16)val;
  }

  __syncthreads();

  // Mpack2[(kb*32 + L)*16 + 2v + s] = M2[k][n], bf16  (k = 8t + l)
  for (int idx = tid; idx < K2STEP * 32 * 16; idx += blockDim.x) {
    const int s  =  idx       & 1;
    const int v  = (idx >> 1) & 7;
    const int L  = (idx >> 4) & 31;
    const int kb =  idx >> 9;
    const int n  = L & 15;
    const int k  = 32 * kb + ((L >= 16) ? 8 : 0) + ((v >= 4) ? 8 : 0) + 2 * v + s;
    float val = 0.f;
    if (n < N_CLASSES && k < K2)
      val = Msh[k * N_CLASSES + n];          // k = 8t+l -> (t*8+l)*10
    mpack[idx] = (__bf16)val;
  }
}

// ---------------------------------------------------------------------------
// Main: WMMA1 z = x*W^T (+bias fused) -> LDS transpose -> per-lane sigmoid/
// path products -> leaf probs bf16 back to LDS -> WMMA2 (leafprobs x M2) ->
// C tile stored directly to out.
// ---------------------------------------------------------------------------
__global__ __launch_bounds__(128)
void sdte_main_kernel(const float* __restrict__ x,
                      const char* __restrict__ ws,
                      float* __restrict__ out,
                      int batch) {
  extern __shared__ char smem[];
  const int tid  = threadIdx.x;
  const int lane = tid & 31;
  const int wave = tid >> 5;

  // Stage Bpack + bias + Mpack2 into LDS (shared by all 4 waves).
  {
    const float4* src = (const float4*)ws;
    float4*       dst = (float4*)smem;
    for (int i = tid; i < WS_TOTAL / 16; i += blockDim.x) dst[i] = src[i];
  }
  __syncthreads();

  const __bf16* bpack = (const __bf16*)(smem + LDS_BPACK);
  const float*  biasL = (const float*)(smem + LDS_BIAS);
  const __bf16* mpk   = (const __bf16*)(smem + LDS_MPACK2);
  char*         slab  = smem + LDS_Z + wave * ZSLAB;     // per-wave scratch
  float*        zs    = (float*)slab;

  const int rowBase  = blockIdx.x * 128 + wave * 32;
  const int halfLane = lane & 15;
  const bool hi      = lane >= 16;

  // Prefetch second half-tile's x rows.
  {
    int prow = rowBase + 16 + halfLane;
    if (prow >= batch) prow = batch - 1;
    __builtin_prefetch(x + (size_t)prow * INPUT_DIM + (hi ? 8 : 0), 0, 0);
  }

  // ================= GEMM1: z = x * W^T =================
  #pragma unroll
  for (int h = 0; h < 2; ++h) {
    const int r0 = rowBase + 16 * h;
    int arow = r0 + halfLane;
    if (arow >= batch) arow = batch - 1;
    const float* xr = x + (size_t)arow * INPUT_DIM + (hi ? 8 : 0);
    v16bf afrag[KSTEP];
    #pragma unroll
    for (int kb = 0; kb < KSTEP; ++kb) {
      const float4 f0 = ((const float4*)(xr + 32 * kb))[0];
      const float4 f1 = ((const float4*)(xr + 32 * kb))[1];
      const float4 f2 = ((const float4*)(xr + 32 * kb + 16))[0];
      const float4 f3 = ((const float4*)(xr + 32 * kb + 16))[1];
      v16bf a;
      a[0]=(__bf16)f0.x; a[1]=(__bf16)f0.y; a[2]=(__bf16)f0.z;  a[3]=(__bf16)f0.w;
      a[4]=(__bf16)f1.x; a[5]=(__bf16)f1.y; a[6]=(__bf16)f1.z;  a[7]=(__bf16)f1.w;
      a[8]=(__bf16)f2.x; a[9]=(__bf16)f2.y; a[10]=(__bf16)f2.z; a[11]=(__bf16)f2.w;
      a[12]=(__bf16)f3.x; a[13]=(__bf16)f3.y; a[14]=(__bf16)f3.z; a[15]=(__bf16)f3.w;
      afrag[kb] = a;
    }
    #pragma unroll
    for (int j = 0; j < NCHUNK; ++j) {
      v8f acc = {0.f, 0.f, 0.f, 0.f, 0.f, 0.f, 0.f, 0.f};
      #pragma unroll
      for (int kb = 0; kb < KSTEP; ++kb) {
        const v16bf b = *(const v16bf*)(bpack + ((j * KSTEP + kb) * 32 + lane) * 16);
        acc = __builtin_amdgcn_wmma_f32_16x16x32_bf16(
            false, afrag[kb], false, b, (short)0, acc, false, false);
      }
      // C layout: VGPR r, lane L -> M = r + (hi?8:0), N = L&15.
      const int col = 16 * j + halfLane;
      if (col < NCOLS) {
        const float bv = biasL[col];
        const int   t  = col / 7;
        const int   nd = col - 7 * t;
        const int   zc = t * 8 + nd;
        const int   mrow = 16 * h + (hi ? 8 : 0);
        #pragma unroll
        for (int r = 0; r < 8; ++r)
          zs[(mrow + r) * ZROW + zc] = acc[r] + bv;
      }
    }
  }
  __syncthreads();

  // ======== per-lane nonlinearity: z -> leaf probabilities (bf16) ========
  unsigned lpbuf[60];                     // 120 leaf probs, bf16-packed pairs
  {
    const float* zr = zs + lane * ZROW;
    #pragma unroll
    for (int t = 0; t < N_TREES; ++t) {
      const float4 z0 = *(const float4*)(zr + t * 8);
      const float4 z1 = *(const float4*)(zr + t * 8 + 4);
      const float p0 = fast_sigmoid(z0.x), p1 = fast_sigmoid(z0.y);
      const float p2 = fast_sigmoid(z0.z), p3 = fast_sigmoid(z0.w);
      const float p4 = fast_sigmoid(z1.x), p5 = fast_sigmoid(z1.y);
      const float p6 = fast_sigmoid(z1.z);
      const float q0 = 1.f - p0, q1 = 1.f - p1, q2 = 1.f - p2, q3 = 1.f - p3;
      const float q4 = 1.f - p4, q5 = 1.f - p5, q6 = 1.f - p6;
      // heap path products (leaf bits b2b1b0, 1 = right = sigmoid(z))
      const float b00 = q0 * q1, b01 = q0 * p1, b10 = p0 * q2, b11 = p0 * p2;
      lpbuf[t * 4 + 0] = pack_bf16(b00 * q3, b00 * p3);
      lpbuf[t * 4 + 1] = pack_bf16(b01 * q4, b01 * p4);
      lpbuf[t * 4 + 2] = pack_bf16(b10 * q5, b10 * p5);
      lpbuf[t * 4 + 3] = pack_bf16(b11 * q6, b11 * p6);
    }
  }
  // Write leaf-prob rows into the (consumed) z slab: lp[row][136 bf16].
  {
    uint4* dst = (uint4*)(slab + lane * LPROW);
    #pragma unroll
    for (int i = 0; i < 15; ++i) {
      uint4 q;
      q.x = lpbuf[4*i+0]; q.y = lpbuf[4*i+1]; q.z = lpbuf[4*i+2]; q.w = lpbuf[4*i+3];
      dst[i] = q;
    }
    uint4 zq; zq.x = zq.y = zq.z = zq.w = 0u;   // K = 120..127 pad
    dst[15] = zq;
  }
  __syncthreads();

  // ================= GEMM2: out = leafprobs * M2 =================
  const int c0 = hi ? 8 : 0;
  #pragma unroll
  for (int h = 0; h < 2; ++h) {
    const int r = 16 * h + halfLane;          // A row within the 32-row slab
    v8f acc = {0.f, 0.f, 0.f, 0.f, 0.f, 0.f, 0.f, 0.f};
    #pragma unroll
    for (int kb = 0; kb < K2STEP; ++kb) {
      union { uint4 u[2]; v16bf v; } af;
      af.u[0] = *(const uint4*)(slab + r * LPROW + (32 * kb + c0) * 2);
      af.u[1] = *(const uint4*)(slab + r * LPROW + (32 * kb + c0 + 16) * 2);
      const v16bf b = *(const v16bf*)(mpk + (kb * 32 + lane) * 16);
      acc = __builtin_amdgcn_wmma_f32_16x16x32_bf16(
          false, af.v, false, b, (short)0, acc, false, false);
    }
    // C layout -> direct global store: row = rowBase+16h+r'+(hi?8:0), col = L&15
    if (halfLane < N_CLASSES) {
      #pragma unroll
      for (int r2 = 0; r2 < 8; ++r2) {
        const int row = rowBase + 16 * h + r2 + (hi ? 8 : 0);
        if (row < batch)
          out[(size_t)row * N_CLASSES + halfLane] = acc[r2];
      }
    }
  }
}

extern "C" void kernel_launch(void* const* d_in, const int* in_sizes, int n_in,
                              void* d_out, int out_size, void* d_ws, size_t ws_size,
                              hipStream_t stream) {
  const float* x      = (const float*)d_in[0];
  const float* params = (const float*)d_in[1];
  float*       out    = (float*)d_out;
  char*        ws     = (char*)d_ws;
  const int batch = in_sizes[0] / INPUT_DIM;

  sdte_prep_kernel<<<1, 256, 0, stream>>>(params, ws);

  const int grid = (batch + 127) / 128;   // 128 rows per block (4 waves x 32)
  sdte_main_kernel<<<grid, 128, LDS_TOTAL, stream>>>(x, ws, out, batch);
}
